// QSVTVelocityField_16544214024240
// MI455X (gfx1250) — compile-verified, matrix operand order checked by reading, not run on hardware
//
#include <hip/hip_runtime.h>
#include <math.h>

// ---------------- static configuration (mirrors reference) ----------------
#define NQ     8
#define NA     2
#define NAMP   1024           // 10 qubits
#define NSEL   4
#define DEG    3
#define NROTS  64             // 4*NQ*LAYERS
#define NOBS   7
#define BATCH  256
#define LATENT 128
#define SIMT   256            // threads per simulate workgroup (8 wave32)

typedef _Float16 v16h __attribute__((ext_vector_type(16)));
typedef float    v8f  __attribute__((ext_vector_type(8)));

// ---------------- complex helpers ----------------
__device__ __forceinline__ float2 cmul(float2 a, float2 b) {
    return make_float2(a.x * b.x - a.y * b.y, a.x * b.y + a.y * b.x);
}
__device__ __forceinline__ float2 cadd(float2 a, float2 b) {
    return make_float2(a.x + b.x, a.y + b.y);
}
__device__ __forceinline__ float fast_sigmoid(float x) {
    // v_exp + v_rcp, no IEEE divide sequence
    return __builtin_amdgcn_rcpf(1.0f + __expf(-x));
}

// =====================================================================
// Kernel 1: sinusoidal embedding   emb[b, 0:64]=cos, emb[b,64:128]=sin
// =====================================================================
__global__ void embed_kernel(const float* __restrict__ t, float* __restrict__ emb) {
    const int b = blockIdx.x, i = threadIdx.x;      // 64 threads
    const float freq = __expf(-9.210340371976184f * (float)i / 64.0f); // ln(1e4)
    const float arg  = t[b] * freq;
    emb[b * LATENT + i]      = __cosf(arg);
    emb[b * LATENT + 64 + i] = __sinf(arg);
}

// =====================================================================
// Kernel 2: WMMA GEMM   out[m,n] = act( sum_k A[m,k]*W[n,k] + bias[n] )
// One wave32 per 16x16 tile. K and ACT are compile-time: the K-loop is
// fully unrolled (K/32 v_wmma_f32_16x16x32_f16 per tile) and fragment
// loads are 128-bit (each lane's 16 f16 elements come from two 8-float
// contiguous, 32B-aligned runs of A / W).
// ACT: 0=none, 1=silu, 2=sigmoid*2pi
// =====================================================================
template <int K, int ACT>
__global__ __launch_bounds__(32)
void wmma_gemm_kernel(const float* __restrict__ A, int lda,
                      const float* __restrict__ W,      // (N,K) row-major
                      const float* __restrict__ bias,
                      float* __restrict__ out, int ldo)
{
    const int lane = threadIdx.x;                   // 0..31
    const int tn = blockIdx.x;                      // tile along N
    const int tm = blockIdx.y;                      // tile along M
    const int row  = tm * 16 + (lane & 15);         // A row for this lane
    const int col  = tn * 16 + (lane & 15);         // B column for this lane
    const int kgrp = (lane < 16) ? 0 : 8;           // K-half split across lanes

    v8f acc = {};
#pragma unroll
    for (int kc = 0; kc < K; kc += 32) {
        const float4* ap = (const float4*)(A + row * lda + kc + kgrp);
        const float4* wp = (const float4*)(W + col * K   + kc + kgrp);
        // elems 0..7 -> K = base+0..7 ; elems 8..15 -> K = base+16..23
        const float4 a0 = ap[0], a1 = ap[1], a2 = ap[4], a3 = ap[5];
        const float4 b0 = wp[0], b1 = wp[1], b2 = wp[4], b3 = wp[5];
        v16h af, bf;
        af[0]=(_Float16)a0.x; af[1]=(_Float16)a0.y; af[2]=(_Float16)a0.z; af[3]=(_Float16)a0.w;
        af[4]=(_Float16)a1.x; af[5]=(_Float16)a1.y; af[6]=(_Float16)a1.z; af[7]=(_Float16)a1.w;
        af[8]=(_Float16)a2.x; af[9]=(_Float16)a2.y; af[10]=(_Float16)a2.z; af[11]=(_Float16)a2.w;
        af[12]=(_Float16)a3.x; af[13]=(_Float16)a3.y; af[14]=(_Float16)a3.z; af[15]=(_Float16)a3.w;
        bf[0]=(_Float16)b0.x; bf[1]=(_Float16)b0.y; bf[2]=(_Float16)b0.z; bf[3]=(_Float16)b0.w;
        bf[4]=(_Float16)b1.x; bf[5]=(_Float16)b1.y; bf[6]=(_Float16)b1.z; bf[7]=(_Float16)b1.w;
        bf[8]=(_Float16)b2.x; bf[9]=(_Float16)b2.y; bf[10]=(_Float16)b2.z; bf[11]=(_Float16)b2.w;
        bf[12]=(_Float16)b3.x; bf[13]=(_Float16)b3.y; bf[14]=(_Float16)b3.z; bf[15]=(_Float16)b3.w;
        acc = __builtin_amdgcn_wmma_f32_16x16x32_f16(
            /*neg_a=*/false, af, /*neg_b=*/false, bf,
            /*c_mod=*/(short)0, acc, /*reuse_a=*/false, /*reuse_b=*/false);
    }

    const int mbase = tm * 16 + ((lane < 16) ? 0 : 8);
#pragma unroll
    for (int r = 0; r < 8; ++r) {
        float v = acc[r] + bias[col];
        if (ACT == 1)      v = v * fast_sigmoid(v);                      // silu
        else if (ACT == 2) v = 6.283185307179586f * fast_sigmoid(v);     // sigmoid*2pi
        out[(mbase + r) * ldo + col] = v;
    }
}

// =====================================================================
// Kernel 3: copy z_t into first half of concat buffer (stride 256)
// =====================================================================
__global__ void concat_copy_kernel(const float* __restrict__ z, float* __restrict__ cat) {
    const int b = blockIdx.x, j = threadIdx.x;      // 128 threads
    cat[b * 256 + j] = z[b * LATENT + j];
}

// =====================================================================
// Kernel 4: build 7 Hermitian 4x4 observables
// =====================================================================
__global__ void hermitian_kernel(const float* __restrict__ A, const float* __restrict__ Bm,
                                 const float* __restrict__ D, float2* __restrict__ Hall)
{
    const int tid = threadIdx.x;                    // 112 threads
    if (tid >= NOBS * 16) return;
    const int w = tid >> 4, e = tid & 15, i = e >> 2, j = e & 3;
    float2 h;
    if (i == j)      h = make_float2((i < 3) ? 2.0f * D[w * 4 + i + 1] : 0.0f, 0.0f);
    else if (i > j) { const int idx = i * (i - 1) / 2 + j; h = make_float2(A[w * 6 + idx],  Bm[w * 6 + idx]); }
    else            { const int idx = j * (j - 1) / 2 + i; h = make_float2(A[w * 6 + idx], -Bm[w * 6 + idx]); }
    Hall[w * 16 + e] = h;
}

// =====================================================================
// Quantum state-vector gate helpers. State lives in LDS as 1024 float2.
// Qubit w has bit position (9 - w) in the amplitude index; ancilla qubits
// 8,9 are the two LSBs (so the ancilla value is idx & 3). Every helper
// ends with a workgroup barrier; all calls are workgroup-uniform.
// =====================================================================
__device__ void ry_gate(float2* st, int q, float c, float s, int anc, int tid) {
    const int pos = 9 - q;
    for (int p = tid; p < 512; p += SIMT) {
        const int low = p & ((1 << pos) - 1);
        const int i0  = ((p >> pos) << (pos + 1)) | low;
        if (anc < 0 || (i0 & 3) == anc) {
            const int i1 = i0 | (1 << pos);
            const float2 a0 = st[i0], a1 = st[i1];
            st[i0] = make_float2(c * a0.x - s * a1.x, c * a0.y - s * a1.y);
            st[i1] = make_float2(s * a0.x + c * a1.x, s * a0.y + c * a1.y);
        }
    }
    __syncthreads();
}

// CRX: on control=1 subspace apply [[c, -i s],[-i s, c]]
__device__ void crx_gate(float2* st, int cq, int tq, float c, float s, int anc, int tid) {
    const int tpos = 9 - tq, cpos = 9 - cq;
    for (int p = tid; p < 512; p += SIMT) {
        const int low = p & ((1 << tpos) - 1);
        const int i0  = ((p >> tpos) << (tpos + 1)) | low;
        if (((i0 >> cpos) & 1) && (anc < 0 || (i0 & 3) == anc)) {
            const int i1 = i0 | (1 << tpos);
            const float2 a0 = st[i0], a1 = st[i1];
            // (-i s)*(x+iy) = (s*y, -s*x)
            st[i0] = make_float2(c * a0.x + s * a1.y, c * a0.y - s * a1.x);
            st[i1] = make_float2(c * a1.x + s * a0.y, c * a1.y - s * a0.x);
        }
    }
    __syncthreads();
}

__device__ void rz_gate(float2* st, int q, float half_th, int tid) {
    const int pos = 9 - q;
    for (int i = tid; i < NAMP; i += SIMT) {
        const float ang = ((i >> pos) & 1) ? half_th : -half_th;
        st[i] = cmul(st[i], make_float2(__cosf(ang), __sinf(ang)));
    }
    __syncthreads();
}

__device__ void cnot_gate(float2* st, int cq, int tq, int tid) {
    const int tpos = 9 - tq, cpos = 9 - cq;
    for (int p = tid; p < 512; p += SIMT) {
        const int low = p & ((1 << tpos) - 1);
        const int i0  = ((p >> tpos) << (tpos + 1)) | low;
        if ((i0 >> cpos) & 1) {
            const int i1 = i0 | (1 << tpos);
            const float2 tmp = st[i0]; st[i0] = st[i1]; st[i1] = tmp;
        }
    }
    __syncthreads();
}

__device__ void pcphase_g(float2* st, float phi, int tid) {
    for (int i = tid; i < NAMP; i += SIMT) {
        const float ang = ((i & 3) == 0) ? phi : -phi;
        st[i] = cmul(st[i], make_float2(__cosf(ang), __sinf(ang)));
    }
    __syncthreads();
}

// sim14 op table for one layer (param index == table position).
// kind 0 = RY(w0), kind 1 = CRX(control=w0, target=w1)
__constant__ unsigned char SIM14_KIND[32] =
    {0,0,0,0,0,0,0,0, 1,1,1,1,1,1,1,1, 0,0,0,0,0,0,0,0, 1,1,1,1,1,1,1,1};
__constant__ unsigned char SIM14_W0[32] =
    {0,1,2,3,4,5,6,7, 7,6,5,4,3,2,1,0, 0,1,2,3,4,5,6,7, 7,0,1,2,3,4,5,6};
__constant__ unsigned char SIM14_W1[32] =
    {0,0,0,0,0,0,0,0, 0,7,6,5,4,3,2,1, 0,0,0,0,0,0,0,0, 6,7,0,1,2,3,4,5};

__device__ void sim14_circ(float2* st, const float* __restrict__ p,
                           int layers, bool adj, int anc, int tid)
{
    const float sgn = adj ? -1.0f : 1.0f;
    for (int l = 0; l < layers; ++l) {
        const int L = adj ? (layers - 1 - l) : l;
        for (int kk = 0; kk < 32; ++kk) {
            const int k  = adj ? (31 - kk) : kk;
            const float th = sgn * p[32 * L + k];
            const float c = __cosf(0.5f * th), s = __sinf(0.5f * th);
            if (SIM14_KIND[k] == 0) ry_gate(st, SIM14_W0[k], c, s, anc, tid);
            else                    crx_gate(st, SIM14_W0[k], SIM14_W1[k], c, s, anc, tid);
        }
    }
}

__device__ void prepare_circ(float2* st, const float* __restrict__ pp, bool adj, int tid) {
    if (!adj) {
        for (int ly = 0; ly < NA; ++ly) {
            for (int qi = 0; qi < NA; ++qi) {
                const float thy = pp[(ly * NA + qi) * 2 + 0];
                const float thz = pp[(ly * NA + qi) * 2 + 1];
                ry_gate(st, NQ + qi, __cosf(0.5f * thy), __sinf(0.5f * thy), -1, tid);
                rz_gate(st, NQ + qi, 0.5f * thz, tid);
            }
            cnot_gate(st, NQ, NQ + 1, tid);
        }
    } else {
        for (int ly = NA - 1; ly >= 0; --ly) {
            cnot_gate(st, NQ, NQ + 1, tid);
            for (int qi = NA - 1; qi >= 0; --qi) {
                const float thy = -pp[(ly * NA + qi) * 2 + 0];
                const float thz = -pp[(ly * NA + qi) * 2 + 1];
                rz_gate(st, NQ + qi, 0.5f * thz, tid);
                ry_gate(st, NQ + qi, __cosf(0.5f * thy), __sinf(0.5f * thy), -1, tid);
            }
        }
    }
}

// =====================================================================
// Kernel 5: full QSVT simulation, one workgroup per batch element.
// State (8 KB) LDS-resident; writes expvals (B, NOBS).
// =====================================================================
__global__ __launch_bounds__(SIMT)
void simulate_kernel(const float* __restrict__ ts,   const float* __restrict__ prep,
                     const float* __restrict__ sig,  const float* __restrict__ qff,
                     const float2* __restrict__ Hall, float* __restrict__ expvals)
{
    __shared__ float2 st[NAMP];
    __shared__ float  red[SIMT];
    const int tid = threadIdx.x;
    const int b   = blockIdx.x;
    const float* tsb = ts + b * (NSEL * NROTS);

    for (int i = tid; i < NAMP; i += SIMT) st[i] = make_float2((i == 0) ? 1.0f : 0.0f, 0.0f);
    __syncthreads();

    pcphase_g(st, sig[0], tid);
    for (int k = 0; k < DEG; ++k) {
        const bool adj = (k & 1);
        prepare_circ(st, prep, false, tid);
        for (int s = 0; s < NSEL; ++s)
            sim14_circ(st, tsb + s * NROTS, 2, adj, s, tid);
        prepare_circ(st, prep, true, tid);
        pcphase_g(st, sig[k + 1], tid);
    }
    sim14_circ(st, qff, 1, false, -1, tid);

    // expectation values <psi| H_w |psi> on qubits (w, w+1)
    for (int w = 0; w < NOBS; ++w) {
        const float2* H = Hall + w * 16;
        const int pa = 9 - w, pb = pa - 1;          // bit positions of qubits w, w+1
        const int low  = tid & ((1 << pb) - 1);     // 256 groups, one per thread
        const int base = ((tid >> pb) << (pb + 2)) | low;
        float2 psi[4];
#pragma unroll
        for (int i = 0; i < 4; ++i)
            psi[i] = st[base + ((i >> 1) << pa) + ((i & 1) << pb)];
        float e = 0.0f;
#pragma unroll
        for (int i = 0; i < 4; ++i) {
            float2 h = make_float2(0.0f, 0.0f);
#pragma unroll
            for (int j = 0; j < 4; ++j) h = cadd(h, cmul(H[i * 4 + j], psi[j]));
            e += psi[i].x * h.x + psi[i].y * h.y;   // Re(conj(psi_i)*h)
        }
        red[tid] = e;
        __syncthreads();
        for (int off = SIMT / 2; off > 0; off >>= 1) {
            if (tid < off) red[tid] += red[tid + off];
            __syncthreads();
        }
        if (tid == 0) expvals[b * NOBS + w] = red[0];
        __syncthreads();
    }
}

// =====================================================================
// Kernel 6: head   out[b,o] = expvals[b,:] . head_w[o,:] + head_b[o]
// =====================================================================
__global__ void head_kernel(const float* __restrict__ ev, const float* __restrict__ hw,
                            const float* __restrict__ hb, float* __restrict__ out)
{
    const int b = blockIdx.x, o = threadIdx.x;      // 128 threads
    float acc = hb[o];
#pragma unroll
    for (int w = 0; w < NOBS; ++w) acc += ev[b * NOBS + w] * hw[o * NOBS + w];
    out[b * LATENT + o] = acc;
}

// =====================================================================
// Launch
// =====================================================================
extern "C" void kernel_launch(void* const* d_in, const int* in_sizes, int n_in,
                              void* d_out, int out_size, void* d_ws, size_t ws_size,
                              hipStream_t stream)
{
    (void)in_sizes; (void)n_in; (void)out_size; (void)ws_size;

    const float* z_t    = (const float*)d_in[0];
    const float* t      = (const float*)d_in[1];
    const float* te_w1  = (const float*)d_in[2];
    const float* te_b1  = (const float*)d_in[3];
    const float* te_w2  = (const float*)d_in[4];
    const float* te_b2  = (const float*)d_in[5];
    const float* ip_w1  = (const float*)d_in[6];
    const float* ip_b1  = (const float*)d_in[7];
    const float* ip_w2  = (const float*)d_in[8];
    const float* ip_b2  = (const float*)d_in[9];
    const float* prep   = (const float*)d_in[10];
    const float* sig    = (const float*)d_in[11];
    const float* qff    = (const float*)d_in[12];
    const float* A_obs  = (const float*)d_in[13];
    const float* B_obs  = (const float*)d_in[14];
    const float* D_obs  = (const float*)d_in[15];
    const float* head_w = (const float*)d_in[16];
    const float* head_b = (const float*)d_in[17];
    float* out = (float*)d_out;

    // workspace carve-out (bytes)
    char* ws = (char*)d_ws;
    float*  tsbuf = (float*)(ws + 0);         // 256*256*4 = 262144
    float*  cat   = (float*)(ws + 262144);    // 256*256*4
    float*  h2    = (float*)(ws + 524288);    // 256*256*4
    float*  emb   = (float*)(ws + 786432);    // 256*128*4
    float*  h1    = (float*)(ws + 917504);    // 256*128*4
    float2* Hall  = (float2*)(ws + 1048576);  // 7*16*8 = 896
    float*  expv  = (float*)(ws + 1049600);   // 256*7*4 = 7168

    // 1) sinusoidal embedding
    embed_kernel<<<BATCH, 64, 0, stream>>>(t, emb);

    // 2) t_emb MLP: h1 = silu(emb @ te_w1^T + te_b1)      (256x128x128)
    wmma_gemm_kernel<LATENT, 1><<<dim3(LATENT / 16, BATCH / 16), 32, 0, stream>>>(
        emb, LATENT, te_w1, te_b1, h1, LATENT);

    // 3) t_emb2 = h1 @ te_w2^T + te_b2, written into cat[:,128:256]
    wmma_gemm_kernel<LATENT, 0><<<dim3(LATENT / 16, BATCH / 16), 32, 0, stream>>>(
        h1, LATENT, te_w2, te_b2, cat + LATENT, 256);

    // 4) cat[:,0:128] = z_t
    concat_copy_kernel<<<BATCH, LATENT, 0, stream>>>(z_t, cat);

    // 5) h2 = silu(cat @ ip_w1^T + ip_b1)                 (256x256x256)
    wmma_gemm_kernel<256, 1><<<dim3(256 / 16, BATCH / 16), 32, 0, stream>>>(
        cat, 256, ip_w1, ip_b1, h2, 256);

    // 6) ts = sigmoid(h2 @ ip_w2^T + ip_b2) * 2pi         (256x256x256)
    wmma_gemm_kernel<256, 2><<<dim3(256 / 16, BATCH / 16), 32, 0, stream>>>(
        h2, 256, ip_w2, ip_b2, tsbuf, 256);

    // 7) observables
    hermitian_kernel<<<1, 112, 0, stream>>>(A_obs, B_obs, D_obs, Hall);

    // 8) quantum simulation: one workgroup per batch element
    simulate_kernel<<<BATCH, SIMT, 0, stream>>>(tsbuf, prep, sig, qff, Hall, expv);

    // 9) head
    head_kernel<<<BATCH, LATENT, 0, stream>>>(expv, head_w, head_b, out);
}
